// GRUDTransformer_12962211300004
// MI455X (gfx1250) — compile-verified, matrix operand order checked by reading, not run on hardware
//
#include <hip/hip_runtime.h>
#include <hip/hip_bf16.h>

// ---- problem constants ----
#define B_ 128
#define S_ 256
#define F_ 40
#define H_ 64
#define D_ 128
#define NH_ 4
#define L_ 2
#define FF_ 2048
#define NEGBIG (-1e9f)

typedef __attribute__((ext_vector_type(16))) __bf16 v16bf;
typedef __attribute__((ext_vector_type(8)))  float  v8f;

__device__ __forceinline__ unsigned short f2bfbits(float f) {
  return __builtin_bit_cast(unsigned short, (__bf16)f);   // native v_cvt, RTNE
}

// K index for element e (0..15) of a 16-bf16 operand, per ISA 7.12.2.
// Elements e=0..7 cover K = k0 + half*8 + e (contiguous), e=8..15 cover
// K = k0 + 16 + half*8 + (e-8) (contiguous) -> two 16B-aligned float4 runs.
__device__ __forceinline__ int kmap(int e, int half) {
  int v = e >> 1, sub = e & 1;
  int kb = (v < 4) ? (v << 1) : (16 + ((v - 4) << 1));
  return kb + half * 8 + sub;
}

__device__ __forceinline__ v8f wmma_bf16(v16bf a, v16bf b, v8f c) {
  return __builtin_amdgcn_wmma_f32_16x16x32_bf16(false, a, false, b, (short)0, c,
                                                 false, false);
}

// Vectorized fp32 global -> bf16 operand tile load (4 x global_load_b128)
__device__ __forceinline__ v16bf load_tile_bf16(const float* __restrict__ p, int ld,
                                                int row, int k0, int half) {
  const float* r0 = p + (size_t)row * ld + k0 + half * 8;
  float4 a0 = *reinterpret_cast<const float4*>(r0);
  float4 a1 = *reinterpret_cast<const float4*>(r0 + 4);
  float4 b0 = *reinterpret_cast<const float4*>(r0 + 16);
  float4 b1 = *reinterpret_cast<const float4*>(r0 + 20);
  v16bf v;
  v[0] = (__bf16)a0.x;  v[1] = (__bf16)a0.y;  v[2] = (__bf16)a0.z;  v[3] = (__bf16)a0.w;
  v[4] = (__bf16)a1.x;  v[5] = (__bf16)a1.y;  v[6] = (__bf16)a1.z;  v[7] = (__bf16)a1.w;
  v[8] = (__bf16)b0.x;  v[9] = (__bf16)b0.y;  v[10] = (__bf16)b0.z; v[11] = (__bf16)b0.w;
  v[12] = (__bf16)b1.x; v[13] = (__bf16)b1.y; v[14] = (__bf16)b1.z; v[15] = (__bf16)b1.w;
  return v;
}

// bf16-bits LDS -> operand tile load (2 x ds_load_b128)
__device__ __forceinline__ v16bf load_tile_lds(const unsigned short* p, int ld,
                                               int row, int k0, int half) {
  const unsigned short* r0 = p + row * ld + k0 + half * 8;
  union { uint4 u[2]; v16bf v; } c;
  c.u[0] = *reinterpret_cast<const uint4*>(r0);
  c.u[1] = *reinterpret_cast<const uint4*>(r0 + 16);
  return c.v;
}

// ---------------------------------------------------------------------------
// time mask -> additive key bias (0 valid / NEGBIG invalid), one per (b,s)
// ---------------------------------------------------------------------------
__global__ void tmask_kernel(const float* __restrict__ mask, float* __restrict__ tb) {
  int b = blockIdx.x, s = threadIdx.x;
  const float* m = mask + ((size_t)b * S_ + s) * F_;
  float acc = 0.f;
  for (int f = 0; f < F_; ++f) acc += m[f];
  tb[b * S_ + s] = (acc > 0.f) ? 0.f : NEGBIG;
}

// ---------------------------------------------------------------------------
// xw = concat(x,mask,delta) @ W_ih^T + b_ih   [B*S, 3H]
// ---------------------------------------------------------------------------
__global__ void xw_kernel(const float* __restrict__ x, const float* __restrict__ mask,
                          const float* __restrict__ delta,
                          const float* __restrict__ w_ih, const float* __restrict__ b_ih,
                          float* __restrict__ xw) {
  int bs = blockIdx.x;
  int g  = threadIdx.x;                 // 0..191
  const float* xr = x + (size_t)bs * F_;
  const float* mr = mask + (size_t)bs * F_;
  const float* dr = delta + (size_t)bs * F_;
  const float* w  = w_ih + (size_t)g * (3 * F_);
  float acc = b_ih[g];
  for (int f = 0; f < F_; ++f)
    acc += xr[f] * w[f] + mr[f] * w[F_ + f] + dr[f] * w[2 * F_ + f];
  xw[(size_t)bs * (3 * H_) + g] = acc;
}

// ---------------------------------------------------------------------------
// GRU scan: one block per batch sample, 64 threads = hidden units; W_hh in LDS
// ---------------------------------------------------------------------------
__global__ __launch_bounds__(64) void gru_kernel(const float* __restrict__ xw,
                                                 const float* __restrict__ w_hh,
                                                 const float* __restrict__ b_hh,
                                                 float* __restrict__ hs) {
  __shared__ float W[3 * H_ * H_];   // 48 KB
  __shared__ float bsh[3 * H_];
  __shared__ float h[H_];
  int b = blockIdx.x, j = threadIdx.x;
  for (int i = j; i < 3 * H_ * H_; i += H_) W[i] = w_hh[i];
  for (int i = j; i < 3 * H_; i += H_) bsh[i] = b_hh[i];
  h[j] = 0.f;
  __syncthreads();
  const float* xwb = xw + (size_t)b * S_ * (3 * H_);
  const float* wr = &W[(size_t)j * H_];
  const float* wz = &W[(size_t)(H_ + j) * H_];
  const float* wn = &W[(size_t)(2 * H_ + j) * H_];
  for (int t = 0; t < S_; ++t) {
    float hr = bsh[j], hz = bsh[H_ + j], hn = bsh[2 * H_ + j];
    for (int k = 0; k < H_; ++k) {
      float hv = h[k];
      hr += wr[k] * hv; hz += wz[k] * hv; hn += wn[k] * hv;
    }
    const float* xt = xwb + (size_t)t * (3 * H_);
    float r  = 1.f / (1.f + __expf(-(xt[j] + hr)));
    float zg = 1.f / (1.f + __expf(-(xt[H_ + j] + hz)));
    float n  = tanhf(xt[2 * H_ + j] + r * hn);
    float hold = h[j];
    float hnew = (1.f - zg) * n + zg * hold;
    __syncthreads();
    h[j] = hnew;
    __syncthreads();
    hs[((size_t)b * S_ + t) * H_ + j] = hnew;
  }
}

// ---------------------------------------------------------------------------
// bf16 WMMA GEMM: C[M,N] = A[M,K] @ W[N,K]^T (+bias) (+resid) (+relu)
// Compile-time K -> fully unrolled; NT output tiles per wave share the A tile.
// grid = (M/16, N/(16*NT)), block = 32 (one wave)
// ---------------------------------------------------------------------------
template <int K, int NT, bool HAS_BIAS, bool HAS_RES, bool RELU>
__global__ __launch_bounds__(32) void gemm_bf16_kernel(
    const float* __restrict__ A, const float* __restrict__ W,
    const float* __restrict__ bias, const float* __restrict__ resid,
    float* __restrict__ C, int N) {
  const int lane = threadIdx.x & 31;
  const int ln = lane & 15, half = lane >> 4;
  const int row0 = blockIdx.x * 16, n0 = blockIdx.y * 16 * NT;
  v8f acc[NT];
#pragma unroll
  for (int t = 0; t < NT; ++t) acc[t] = (v8f){0.f, 0.f, 0.f, 0.f, 0.f, 0.f, 0.f, 0.f};
#pragma unroll
  for (int k0 = 0; k0 < K; k0 += 32) {
    v16bf a = load_tile_bf16(A, K, row0 + ln, k0, half);
#pragma unroll
    for (int t = 0; t < NT; ++t) {
      v16bf b = load_tile_bf16(W, K, n0 + t * 16 + ln, k0, half);
      acc[t] = wmma_bf16(a, b, acc[t]);
    }
  }
#pragma unroll
  for (int t = 0; t < NT; ++t) {
    const int n = n0 + t * 16 + ln;
    const float bv = HAS_BIAS ? bias[n] : 0.f;
#pragma unroll
    for (int r = 0; r < 8; ++r) {
      int m = row0 + r + 8 * half;
      float v = acc[t][r] + bv;
      if (HAS_RES) v += resid[(size_t)m * N + n];
      if (RELU) v = fmaxf(v, 0.f);
      C[(size_t)m * N + n] = v;
    }
  }
}

// ---------------------------------------------------------------------------
// Fused attention: one wave per (b, head, 16-query strip).
// scores (WMMA, K=32) -> softmax in LDS -> P@V (WMMA, K=256)
// ---------------------------------------------------------------------------
__global__ __launch_bounds__(32) void attn_kernel(const float* __restrict__ qkv,
                                                  const float* __restrict__ tb,
                                                  float* __restrict__ O) {
  __shared__ float sc[16 * S_];           // 16 KB scores
  __shared__ unsigned short P[16 * S_];   // 8 KB bf16 probabilities
  const int id = blockIdx.x;
  const int strip = id & 15;
  const int hh = (id >> 4) & (NH_ - 1);
  const int b = id >> 6;
  const int lane = threadIdx.x & 31;
  const int ln = lane & 15, half = lane >> 4;
  const int row0 = strip * 16;
  const float scale = 0.17677669529663687f;   // 1/sqrt(32)
  const float* qh = qkv + (size_t)b * S_ * (3 * D_) + hh * 32;   // q plane
  const float* kh = qh + D_;                                     // k plane
  const float* vh = qh + 2 * D_;                                 // v plane

  // Q strip (16 x 32) held in registers, reused for all 16 key tiles
  v16bf a = load_tile_bf16(qh, 3 * D_, row0 + ln, 0, half);

#pragma unroll 4
  for (int jt = 0; jt < 16; ++jt) {
    v16bf bm = load_tile_bf16(kh, 3 * D_, jt * 16 + ln, 0, half);
    v8f c = {0.f, 0.f, 0.f, 0.f, 0.f, 0.f, 0.f, 0.f};
    c = wmma_bf16(a, bm, c);
    int j = jt * 16 + ln;
    float bias = tb[b * S_ + j];
#pragma unroll
    for (int r = 0; r < 8; ++r)
      sc[(r + 8 * half) * S_ + j] = c[r] * scale + bias;
  }
  __syncthreads();

  if (lane < 16) {   // one lane per query row: softmax over 256 keys
    float mx = -3.4e38f;
    for (int j = 0; j < S_; ++j) mx = fmaxf(mx, sc[lane * S_ + j]);
    float sum = 0.f;
    for (int j = 0; j < S_; ++j) {
      float e = __expf(sc[lane * S_ + j] - mx);
      sum += e;
      sc[lane * S_ + j] = e;
    }
    float inv = 1.f / sum;
    for (int j = 0; j < S_; ++j)
      P[lane * S_ + j] = f2bfbits(sc[lane * S_ + j] * inv);
  }
  __syncthreads();

  // out(16 x 32) = P(16 x 256) @ V(256 x 32)
#pragma unroll
  for (int nt = 0; nt < 2; ++nt) {
    v8f acc = {0.f, 0.f, 0.f, 0.f, 0.f, 0.f, 0.f, 0.f};
#pragma unroll
    for (int k0 = 0; k0 < S_; k0 += 32) {
      v16bf pa = load_tile_lds(P, S_, ln, k0, half);
      v16bf vb;
#pragma unroll
      for (int e = 0; e < 16; ++e)
        vb[e] = (__bf16)vh[(size_t)(k0 + kmap(e, half)) * (3 * D_) + nt * 16 + ln];
      acc = wmma_bf16(pa, vb, acc);
    }
#pragma unroll
    for (int r = 0; r < 8; ++r)
      O[((size_t)b * S_ + row0 + r + 8 * half) * D_ + hh * 32 + nt * 16 + ln] = acc[r];
  }
}

// ---------------------------------------------------------------------------
// Fused FFN: one block (8 waves) per 16-row strip. Hidden 16x2048 ReLU
// activations live in LDS as bf16 (64 KB). T = relu(Z@W1^T+b1)@W2^T+b2 + Z
// ---------------------------------------------------------------------------
__global__ __launch_bounds__(256) void ffn_kernel(const float* __restrict__ Z,
                                                  const float* __restrict__ W1,
                                                  const float* __restrict__ b1,
                                                  const float* __restrict__ W2,
                                                  const float* __restrict__ b2,
                                                  float* __restrict__ T) {
  __shared__ unsigned short hbuf[16 * FF_];   // 64 KB bf16
  const int row0 = blockIdx.x * 16;
  const int lane = threadIdx.x & 31;
  const int wave = threadIdx.x >> 5;
  const int ln = lane & 15, half = lane >> 4;

  // A strip (16 x 128) cached in registers: 4 K-step operands
  v16bf areg[4];
#pragma unroll
  for (int kk = 0; kk < 4; ++kk)
    areg[kk] = load_tile_bf16(Z, D_, row0 + ln, kk * 32, half);

  // Phase 1: hidden = relu(Z @ W1^T + b1), each wave covers 256 of 2048 cols
#pragma unroll 2
  for (int tt = 0; tt < 16; ++tt) {
    const int n0 = wave * 256 + tt * 16;
    v8f acc = {0.f, 0.f, 0.f, 0.f, 0.f, 0.f, 0.f, 0.f};
#pragma unroll
    for (int kk = 0; kk < 4; ++kk) {
      v16bf bm = load_tile_bf16(W1, D_, n0 + ln, kk * 32, half);
      acc = wmma_bf16(areg[kk], bm, acc);
    }
    const int n = n0 + ln;
    const float bv = b1[n];
#pragma unroll
    for (int r = 0; r < 8; ++r)
      hbuf[(r + 8 * half) * FF_ + n] = f2bfbits(fmaxf(acc[r] + bv, 0.f));
  }
  __syncthreads();

  // Phase 2: out = hidden @ W2^T + b2 + Z ; one 16x16 tile per wave, K=2048
  {
    const int n0 = wave * 16;
    const float* W2row = W2 + (size_t)(n0 + ln) * FF_;
    v8f acc = {0.f, 0.f, 0.f, 0.f, 0.f, 0.f, 0.f, 0.f};
#pragma unroll 4
    for (int k0 = 0; k0 < FF_; k0 += 32) {
      __builtin_prefetch(W2row + k0 + 32);
      v16bf pa = load_tile_lds(hbuf, FF_, ln, k0, half);
      v16bf bm = load_tile_bf16(W2, FF_, n0 + ln, k0, half);
      acc = wmma_bf16(pa, bm, acc);
    }
    const int n = n0 + ln;
    const float bv = b2[n];
#pragma unroll
    for (int r = 0; r < 8; ++r) {
      int m = row0 + r + 8 * half;
      T[(size_t)m * D_ + n] = acc[r] + bv + Z[(size_t)m * D_ + n];
    }
  }
}

// ---------------------------------------------------------------------------
// Row LayerNorm over D=128: one block of 128 threads per row
// ---------------------------------------------------------------------------
__global__ __launch_bounds__(128) void ln_kernel(const float* __restrict__ T,
                                                 const float* __restrict__ sg,
                                                 const float* __restrict__ bg,
                                                 float* __restrict__ Z) {
  __shared__ float red[D_];
  const int row = blockIdx.x, d = threadIdx.x;
  float v = T[(size_t)row * D_ + d];
  red[d] = v;
  __syncthreads();
  for (int o = 64; o > 0; o >>= 1) {
    if (d < o) red[d] += red[d + o];
    __syncthreads();
  }
  float m = red[0] * (1.f / D_);
  __syncthreads();
  float c = v - m;
  red[d] = c * c;
  __syncthreads();
  for (int o = 64; o > 0; o >>= 1) {
    if (d < o) red[d] += red[d + o];
    __syncthreads();
  }
  float var = red[0] * (1.f / D_);
  Z[(size_t)row * D_ + d] = c * rsqrtf(var + 1e-5f) * sg[d] + bg[d];
}

// ---------------------------------------------------------------------------
// Attention pooling: scores -> masked softmax over S -> weighted sum
// ---------------------------------------------------------------------------
__global__ __launch_bounds__(256) void pool_kernel(const float* __restrict__ Z,
                                                   const float* __restrict__ tb,
                                                   const float* __restrict__ sw,
                                                   const float* __restrict__ sb,
                                                   float* __restrict__ pooled) {
  __shared__ float sc[S_];
  __shared__ float red[S_];
  const int b = blockIdx.x, s = threadIdx.x;
  float acc = sb[0];
  const float* zr = Z + ((size_t)b * S_ + s) * D_;
  for (int d = 0; d < D_; ++d) acc += zr[d] * sw[d];
  if (tb[b * S_ + s] != 0.f) acc = NEGBIG;
  red[s] = acc;
  __syncthreads();
  for (int o = 128; o > 0; o >>= 1) {
    if (s < o) red[s] = fmaxf(red[s], red[s + o]);
    __syncthreads();
  }
  float mx = red[0];
  __syncthreads();
  float e = __expf(acc - mx);
  sc[s] = e;
  red[s] = e;
  __syncthreads();
  for (int o = 128; o > 0; o >>= 1) {
    if (s < o) red[s] += red[s + o];
    __syncthreads();
  }
  float inv = 1.f / red[0];
  __syncthreads();
  sc[s] = sc[s] * inv;
  __syncthreads();
  if (s < D_) {
    float p = 0.f;
    for (int j = 0; j < S_; ++j) p += sc[j] * Z[((size_t)b * S_ + j) * D_ + s];
    pooled[(size_t)b * D_ + s] = p;
  }
}

// ---------------------------------------------------------------------------
// Routed heads: 1-of-3 head weights per sample
// ---------------------------------------------------------------------------
__global__ __launch_bounds__(32) void heads_kernel(const float* __restrict__ pooled,
                                                   const float* __restrict__ reg_w,
                                                   const float* __restrict__ reg_b,
                                                   const float* __restrict__ bin_w,
                                                   const float* __restrict__ bin_b,
                                                   const int* __restrict__ wid,
                                                   float* __restrict__ out) {
  const int b = blockIdx.x, t = threadIdx.x;
  const int w = wid[b];
  const float* p = pooled + (size_t)b * D_;
  if (t < 8) {
    const float* wr = reg_w + ((size_t)w * 8 + t) * D_;
    float acc = reg_b[w * 8 + t];
    for (int d = 0; d < D_; ++d) acc += wr[d] * p[d];
    out[b * 8 + t] = acc;
  } else if (t == 8) {
    const float* wb = bin_w + (size_t)w * D_;
    float acc = bin_b[w];
    for (int d = 0; d < D_; ++d) acc += wb[d] * p[d];
    out[B_ * 8 + b] = acc;
  }
}

// ---------------------------------------------------------------------------
extern "C" void kernel_launch(void* const* d_in, const int* in_sizes, int n_in,
                              void* d_out, int out_size, void* d_ws, size_t ws_size,
                              hipStream_t stream) {
  (void)in_sizes; (void)n_in; (void)out_size; (void)ws_size;
  const float* x        = (const float*)d_in[0];
  const float* mask     = (const float*)d_in[1];
  const float* delta    = (const float*)d_in[2];
  const float* gru_w_ih = (const float*)d_in[3];
  const float* gru_w_hh = (const float*)d_in[4];
  const float* gru_b_ih = (const float*)d_in[5];
  const float* gru_b_hh = (const float*)d_in[6];
  const float* dm_w     = (const float*)d_in[7];
  const float* dm_b     = (const float*)d_in[8];
  const float* in_proj_w  = (const float*)d_in[9];
  const float* in_proj_b  = (const float*)d_in[10];
  const float* out_proj_w = (const float*)d_in[11];
  const float* out_proj_b = (const float*)d_in[12];
  const float* lin1_w   = (const float*)d_in[13];
  const float* lin1_b   = (const float*)d_in[14];
  const float* lin2_w   = (const float*)d_in[15];
  const float* lin2_b   = (const float*)d_in[16];
  const float* ln1_s    = (const float*)d_in[17];
  const float* ln1_b    = (const float*)d_in[18];
  const float* ln2_s    = (const float*)d_in[19];
  const float* ln2_b    = (const float*)d_in[20];
  const float* score_w  = (const float*)d_in[21];
  const float* score_b  = (const float*)d_in[22];
  const float* reg_w    = (const float*)d_in[23];
  const float* reg_b    = (const float*)d_in[24];
  const float* bin_w    = (const float*)d_in[25];
  const float* bin_b    = (const float*)d_in[26];
  const int*   wid      = (const int*)d_in[27];

  const int M = B_ * S_;   // 32768 token rows
  float* ws  = (float*)d_ws;
  float* QKV = ws;                              // 32768*384 (also hosts XW early)
  float* XW  = ws;                              // 32768*192, lifetime-disjoint w/ QKV
  float* HS  = QKV + (size_t)M * 3 * D_;        // 32768*64
  float* Z   = HS + (size_t)M * H_;             // 32768*128
  float* T   = Z + (size_t)M * D_;              // 32768*128
  float* O   = T + (size_t)M * D_;              // 32768*128
  float* TB  = O + (size_t)M * D_;              // 32768
  float* PO  = TB + M;                          // 128*128

  tmask_kernel<<<B_, S_, 0, stream>>>(mask, TB);
  xw_kernel<<<M, 3 * H_, 0, stream>>>(x, mask, delta, gru_w_ih, gru_b_ih, XW);
  gru_kernel<<<B_, H_, 0, stream>>>(XW, gru_w_hh, gru_b_hh, HS);

  // z = hs @ dm_w^T + dm_b   (K=64, 4 N-tiles/wave)
  gemm_bf16_kernel<H_, 4, true, false, false>
      <<<dim3(M / 16, D_ / 64), 32, 0, stream>>>(HS, dm_w, dm_b, nullptr, Z, D_);

  for (int l = 0; l < L_; ++l) {
    gemm_bf16_kernel<D_, 4, true, false, false>
        <<<dim3(M / 16, (3 * D_) / 64), 32, 0, stream>>>(
            Z, in_proj_w + (size_t)l * 3 * D_ * D_, in_proj_b + (size_t)l * 3 * D_,
            nullptr, QKV, 3 * D_);
    attn_kernel<<<B_ * NH_ * (S_ / 16), 32, 0, stream>>>(QKV, TB, O);
    gemm_bf16_kernel<D_, 4, true, true, false>
        <<<dim3(M / 16, D_ / 64), 32, 0, stream>>>(
            O, out_proj_w + (size_t)l * D_ * D_, out_proj_b + (size_t)l * D_,
            Z, T, D_);
    ln_kernel<<<M, D_, 0, stream>>>(T, ln1_s + l * D_, ln1_b + l * D_, Z);
    ffn_kernel<<<M / 16, 256, 0, stream>>>(
        Z, lin1_w + (size_t)l * FF_ * D_, lin1_b + (size_t)l * FF_,
        lin2_w + (size_t)l * D_ * FF_, lin2_b + (size_t)l * D_, T);
    ln_kernel<<<M, D_, 0, stream>>>(T, ln2_s + l * D_, ln2_b + l * D_, Z);
  }

  pool_kernel<<<B_, S_, 0, stream>>>(Z, TB, score_w, score_b, PO);
  heads_kernel<<<B_, 32, 0, stream>>>(PO, reg_w, reg_b, bin_w, bin_b, wid,
                                      (float*)d_out);
}
